// Block_81913616269542
// MI455X (gfx1250) — matrix operand order repaired and checked by reading, not compiled, not run on hardware
//
#include <hip/hip_runtime.h>
#include <hip/hip_bf16.h>
#include <math.h>

// ---------------------------------------------------------------------------
// Types for CDNA5 WMMA (gfx1250, wave32)
// ---------------------------------------------------------------------------
typedef __bf16 bf16_t;
typedef __attribute__((ext_vector_type(16))) __bf16 v16bf;
typedef __attribute__((ext_vector_type(8)))  float  v8f;

union FragBF {
    v16bf v;
    uint4 q[2];
    unsigned int ui[8];
};

__device__ __forceinline__ v8f wmma_bf16(v16bf a, v16bf b, v8f c) {
    // D = A(16x32 bf16) * B(32x16 bf16) + C(16x16 f32)
    return __builtin_amdgcn_wmma_f32_16x16x32_bf16(
        /*neg_a=*/false, a, /*neg_b=*/false, b,
        /*c_mod=*/(short)0, c, /*reuse_a=*/false, /*reuse_b=*/false);
}

// Async global->LDS 16B copy (per enabled lane), tracked by ASYNCcnt.
__device__ __forceinline__ void async_copy_b128(unsigned lds_addr,
                                                const void* gptr) {
    asm volatile("global_load_async_to_lds_b128 %0, %1, off"
                 :
                 : "v"(lds_addr), "v"((unsigned long long)(size_t)gptr)
                 : "memory");
}

#if __has_builtin(__builtin_amdgcn_s_wait_asynccnt)
#define WAIT_ASYNC(n) __builtin_amdgcn_s_wait_asynccnt(n)
#else
#define WAIT_ASYNC(n) asm volatile("s_wait_asynccnt %0" ::"i"(n) : "memory")
#endif

// A fragment: rows m=0..15 (A already offset to row0), cols k0..k0+31.
// Layout: lane m=l&15, half hi=l>>4;
//   vgpr0..7 hold k = 16*hi + 2j,2j+1 (contiguous, matches B loaders)
__device__ __forceinline__ v16bf load_a_frag(const bf16_t* __restrict__ A,
                                             int lda, int k0) {
    int lane = threadIdx.x & 31;
    int m = lane & 15, hi = lane >> 4;
    const bf16_t* p = A + (size_t)m * lda + k0 + 16 * hi;
    FragBF f;
    f.q[0] = *(const uint4*)(p);
    f.q[1] = *(const uint4*)(p + 8);
    return f.v;
}

// B fragment from Bt stored [N][K] row-major: B(k,n) = Bt[n*ldb + k].
__device__ __forceinline__ v16bf load_b_fragT(const bf16_t* __restrict__ Bt,
                                              int ldb, int k0) {
    int lane = threadIdx.x & 31;
    int n = lane & 15, hi = lane >> 4;
    const bf16_t* p = Bt + (size_t)n * ldb + k0 + 16 * hi;
    FragBF f;
    f.q[0] = *(const uint4*)(p);
    f.q[1] = *(const uint4*)(p + 8);
    return f.v;
}

// B fragment from row-major K x N matrix: B(k,n) = P[k*ld + n] (strided pairs).
__device__ __forceinline__ v16bf load_b_rowmajor(const bf16_t* __restrict__ P,
                                                 int ld) {
    int lane = threadIdx.x & 31;
    int n = lane & 15, hi = lane >> 4;
    FragBF f;
#pragma unroll
    for (int j = 0; j < 8; ++j) {
        int k = 16 * hi + 2 * j;
        unsigned lo = *(const unsigned short*)(P + (size_t)k * ld + n);
        unsigned hh = *(const unsigned short*)(P + (size_t)(k + 1) * ld + n);
        f.ui[j] = lo | (hh << 16);
    }
    return f.v;
}

// ---------------------------------------------------------------------------
// Weight convert + transpose: src f32 [h][k][d] -> dst bf16 [h][d][k]
// ---------------------------------------------------------------------------
__global__ __launch_bounds__(256) void convert_transpose(
    const float* __restrict__ src, bf16_t* __restrict__ dst,
    int H, int K, int D) {
    size_t total = (size_t)H * K * D;
    size_t kd = (size_t)K * D;
    for (size_t idx = (size_t)blockIdx.x * blockDim.x + threadIdx.x;
         idx < total; idx += (size_t)gridDim.x * blockDim.x) {
        size_t h = idx / kd;
        size_t rem = idx - h * kd;
        size_t k = rem / D, d = rem - k * D;
        dst[(h * D + d) * K + k] = (bf16_t)src[idx];
    }
}

// ---------------------------------------------------------------------------
// RMSNorm: one row per block, f32 in -> bf16 out (scaled by g)
// ---------------------------------------------------------------------------
__global__ __launch_bounds__(256) void rmsnorm_kernel(
    const float* __restrict__ x, const float* __restrict__ g,
    bf16_t* __restrict__ out, int E) {
    __shared__ float red[8];
    __shared__ float totalS;
    int row = blockIdx.x;
    const float* xr = x + (size_t)row * E;
    float ss = 0.f;
    for (int i = threadIdx.x; i < E; i += blockDim.x) {
        float v = xr[i];
        ss += v * v;
    }
#pragma unroll
    for (int off = 16; off > 0; off >>= 1) ss += __shfl_xor(ss, off, 32);
    if ((threadIdx.x & 31) == 0) red[threadIdx.x >> 5] = ss;
    __syncthreads();
    if (threadIdx.x == 0) {
        float t = 0.f;
#pragma unroll
        for (int i = 0; i < 8; ++i) t += red[i];
        totalS = t;
    }
    __syncthreads();
    float inv = rsqrtf(totalS / (float)E + 1e-8f);
    for (int i = threadIdx.x; i < E; i += blockDim.x)
        out[(size_t)row * E + i] = (bf16_t)(xr[i] * inv * g[i]);
}

// ---------------------------------------------------------------------------
// WMMA GEMM with double-buffered async-LDS staging:
//   C[M,N] = act(A[M,K] @ B + bias) + resid
//   A: bf16 row-major [M,K];  Bt: bf16 [N,K] (pre-transposed weights)
//   block = 256 threads (8 wave32), block tile 128x64, wave tile 32x32 (2x2)
//   Per k-stage (32): A tile 128x32 (8KB) + B tile 64x32 (4KB) staged into
//   LDS via global_load_async_to_lds_b128 (3 issues/wave/stage, ASYNCcnt).
// ---------------------------------------------------------------------------
__global__ __launch_bounds__(256) void gemm_wmma(
    const bf16_t* __restrict__ A, const bf16_t* __restrict__ Bt,
    const float* __restrict__ bias, const float* __restrict__ resid,
    float* __restrict__ outF, bf16_t* __restrict__ outB,
    int M, int N, int K, int act) {
    __shared__ __align__(16) bf16_t sA[2][128 * 32];  // 2 x 8KB
    __shared__ __align__(16) bf16_t sB[2][64 * 32];   // 2 x 4KB

    int t = threadIdx.x;
    int wave = t >> 5, lane = t & 31;
    int row0 = blockIdx.x * 128, col0 = blockIdx.y * 64;
    int wm = (wave & 3) * 32, wn = (wave >> 2) * 32;
    int n = lane & 15, hi = lane >> 4;

    // Per-thread async-copy assignment: 16B chunks.
    //   A tile: 128 rows x 4 chunks = 512 chunks -> chunks t and t+256
    //   B tile:  64 rows x 4 chunks = 256 chunks -> chunk t
    int aR0 = t >> 2, aC = (t & 3) * 8;
    int aR1 = aR0 + 64;
    int bR = t >> 2;
    const bf16_t* gA0 = A + (size_t)(row0 + aR0) * K + aC;
    const bf16_t* gA1 = A + (size_t)(row0 + aR1) * K + aC;
    const bf16_t* gB  = Bt + (size_t)(col0 + bR) * K + aC;
    unsigned lA0[2], lA1[2], lB[2];
#pragma unroll
    for (int bfi = 0; bfi < 2; ++bfi) {
        lA0[bfi] = (unsigned)(size_t)&sA[bfi][aR0 * 32 + aC];
        lA1[bfi] = (unsigned)(size_t)&sA[bfi][aR1 * 32 + aC];
        lB[bfi]  = (unsigned)(size_t)&sB[bfi][bR * 32 + aC];
    }

    auto issue_stage = [&](int stage) {
        int buf = stage & 1;
        size_t k0 = (size_t)stage * 32;
        async_copy_b128(lA0[buf], gA0 + k0);
        async_copy_b128(lA1[buf], gA1 + k0);
        async_copy_b128(lB[buf], gB + k0);
    };

    v8f acc00 = 0.f, acc01 = 0.f, acc10 = 0.f, acc11 = 0.f;

    int nstages = K / 32;
    issue_stage(0);
    for (int s = 0; s < nstages; ++s) {
        if (s + 1 < nstages) {
            issue_stage(s + 1);
            WAIT_ASYNC(3);  // in-order: stage s complete, s+1 in flight
        } else {
            WAIT_ASYNC(0);
        }
        __syncthreads();
        const bf16_t* tA = &sA[s & 1][0];
        const bf16_t* tB = &sB[s & 1][0];
        v16bf a0 = load_a_frag(tA + wm * 32, 32, 0);
        v16bf a1 = load_a_frag(tA + (wm + 16) * 32, 32, 0);
        v16bf b0 = load_b_fragT(tB + wn * 32, 32, 0);
        v16bf b1 = load_b_fragT(tB + (wn + 16) * 32, 32, 0);
        acc00 = wmma_bf16(a0, b0, acc00);
        acc01 = wmma_bf16(a0, b1, acc01);
        acc10 = wmma_bf16(a1, b0, acc10);
        acc11 = wmma_bf16(a1, b1, acc11);
        __syncthreads();  // buffer s&1 is overwritten at stage s+2
    }

    v8f* accs[2][2] = {{&acc00, &acc01}, {&acc10, &acc11}};
#pragma unroll
    for (int i = 0; i < 2; ++i)
#pragma unroll
        for (int j = 0; j < 2; ++j) {
            v8f a = *accs[i][j];
#pragma unroll
            for (int v = 0; v < 8; ++v) {
                int m = row0 + wm + i * 16 + v + 8 * hi;
                int c = col0 + wn + j * 16 + n;
                float val = a[v];
                if (bias) val += bias[c];
                if (act == 1)  // exact GELU
                    val = 0.5f * val * (1.0f + erff(val * 0.70710678118654752f));
                size_t o = (size_t)m * N + c;
                if (resid) val += resid[o];
                if (outF) outF[o] = val;
                if (outB) outB[o] = (bf16_t)val;
            }
        }
}

// ---------------------------------------------------------------------------
// Causal attention: one wave per (b, h, 16-query-row tile).
//   q,k,v bf16 laid out [b, t, h*DH + d].  Scores tile kept in 64KB LDS (bf16),
//   row max/sum via intra-wave shfl butterflies (wave32).  o out in bf16.
// ---------------------------------------------------------------------------
__global__ __launch_bounds__(32) void attn_kernel(
    const bf16_t* __restrict__ q, const bf16_t* __restrict__ k,
    const bf16_t* __restrict__ v, bf16_t* __restrict__ o,
    int Bb, int T, int Hh, int E, int DH) {
    __shared__ __align__(16) bf16_t sc[16 * 2048];  // 64KB score/prob tile

    int qt = blockIdx.x, h = blockIdx.y, b = blockIdx.z;
    int r0 = qt * 16;
    int lane = threadIdx.x & 31;
    int n = lane & 15, hi = lane >> 4;

    const bf16_t* qp = q + ((size_t)(b * T + r0)) * E + h * DH;
    const bf16_t* kp = k + ((size_t)(b * T)) * E + h * DH;
    const bf16_t* vp = v + ((size_t)(b * T)) * E + h * DH;

    int Kcap = (r0 + 16 + 31) & ~31;
    if (Kcap > T) Kcap = T;
    const float scale = 0.088388347648318447f;  // 1/sqrt(128)

    float rmax[8];
#pragma unroll
    for (int j = 0; j < 8; ++j) rmax[j] = -INFINITY;

    // ---- pass 1: S = QK^T * scale, causal mask, bf16 store to LDS, row max
    for (int s0 = 0; s0 < Kcap; s0 += 16) {
        v8f accs = 0.f;
#pragma unroll
        for (int c = 0; c < 4; ++c) {  // DH=128 in 4 chunks of 32
            v16bf a = load_a_frag(qp, E, c * 32);
            v16bf bb = load_b_fragT(kp + (size_t)s0 * E, E, c * 32);
            accs = wmma_bf16(a, bb, accs);
        }
#pragma unroll
        for (int j = 0; j < 8; ++j) {
            int m = j + 8 * hi;
            int col = s0 + n;
            float sv = accs[j] * scale;
            if (col > r0 + m) sv = -INFINITY;
            if (sv > rmax[j]) rmax[j] = sv;
            sc[m * 2048 + col] = (bf16_t)sv;
        }
    }

    // row max across the 16 lanes of each half, then swap halves
    float rowmaxAll[16];
#pragma unroll
    for (int j = 0; j < 8; ++j) {
        float mv = rmax[j];
        mv = fmaxf(mv, __shfl_xor(mv, 1, 32));
        mv = fmaxf(mv, __shfl_xor(mv, 2, 32));
        mv = fmaxf(mv, __shfl_xor(mv, 4, 32));
        mv = fmaxf(mv, __shfl_xor(mv, 8, 32));
        float ov = __shfl_xor(mv, 16, 32);  // other half's row j+8*(1-hi)
        rowmaxAll[j + 8 * hi] = mv;
        rowmaxAll[j + 8 * (1 - hi)] = ov;
    }
    __syncthreads();

    // ---- pass 2: P = exp(S - max), row sums, P back to LDS as bf16
    float part[16];
#pragma unroll
    for (int m = 0; m < 16; ++m) part[m] = 0.f;
    for (int col = lane; col < Kcap; col += 32) {
#pragma unroll
        for (int m = 0; m < 16; ++m) {
            float sv = (float)sc[m * 2048 + col];
            float p = __expf(sv - rowmaxAll[m]);  // exp(-inf - max) = 0
            part[m] += p;
            sc[m * 2048 + col] = (bf16_t)p;
        }
    }
    float rowsumAll[16];
#pragma unroll
    for (int m = 0; m < 16; ++m) {
        float p = part[m];
        p += __shfl_xor(p, 1, 32);
        p += __shfl_xor(p, 2, 32);
        p += __shfl_xor(p, 4, 32);
        p += __shfl_xor(p, 8, 32);
        p += __shfl_xor(p, 16, 32);
        rowsumAll[m] = p;
    }
    __syncthreads();

    // ---- pass 3: O = P @ V (A-fragments from LDS), normalize, store bf16
#pragma unroll
    for (int nt = 0; nt < 8; ++nt) {  // DH/16 column tiles
        v8f acco = 0.f;
        for (int s0 = 0; s0 < Kcap; s0 += 32) {
            v16bf a = load_a_frag((const bf16_t*)sc, 2048, s0);
            v16bf bb = load_b_rowmajor(vp + (size_t)s0 * E + nt * 16, E);
            acco = wmma_bf16(a, bb, acco);
        }
#pragma unroll
        for (int j = 0; j < 8; ++j) {
            int m = j + 8 * hi;
            float ov = acco[j] / rowsumAll[m];
            o[((size_t)(b * T + r0 + m)) * E + h * DH + nt * 16 + n] = (bf16_t)ov;
        }
    }
}

// ---------------------------------------------------------------------------
// Host orchestration
// ---------------------------------------------------------------------------
extern "C" void kernel_launch(void* const* d_in, const int* in_sizes, int n_in,
                              void* d_out, int out_size, void* d_ws, size_t ws_size,
                              hipStream_t stream) {
    const int E = 2048, H = 16, DH = 128, Bc = 2, Tc = 2048;
    const int M = Bc * Tc;  // 4096
    const int FF = 4 * E;   // 8192

    const float* x  = (const float*)d_in[0];
    const float* wq = (const float*)d_in[1];
    const float* bq = (const float*)d_in[2];
    const float* wk = (const float*)d_in[3];
    const float* bk = (const float*)d_in[4];
    const float* wv = (const float*)d_in[5];
    const float* bv = (const float*)d_in[6];
    const float* wo = (const float*)d_in[7];
    const float* bo = (const float*)d_in[8];
    const float* w1 = (const float*)d_in[9];
    const float* b1 = (const float*)d_in[10];
    const float* w2 = (const float*)d_in[11];
    const float* b2 = (const float*)d_in[12];
    const float* g1 = (const float*)d_in[13];
    const float* g2 = (const float*)d_in[14];

    // Workspace carve (256B aligned)
    char* base = (char*)d_ws;
    size_t off = 0;
    auto take = [&](size_t bytes) -> void* {
        off = (off + 255) & ~(size_t)255;
        void* p = base + off;
        off += bytes;
        return p;
    };
    bf16_t* h1  = (bf16_t*)take((size_t)M * E * 2);
    bf16_t* wqT = (bf16_t*)take((size_t)E * E * 2);
    bf16_t* wkT = (bf16_t*)take((size_t)E * E * 2);
    bf16_t* wvT = (bf16_t*)take((size_t)E * E * 2);
    bf16_t* woT = (bf16_t*)take((size_t)E * E * 2);
    bf16_t* w1T = (bf16_t*)take((size_t)E * FF * 2);
    bf16_t* w2T = (bf16_t*)take((size_t)FF * E * 2);
    bf16_t* qb  = (bf16_t*)take((size_t)M * E * 2);
    bf16_t* kb  = (bf16_t*)take((size_t)M * E * 2);
    bf16_t* vb  = (bf16_t*)take((size_t)M * E * 2);
    bf16_t* ob  = (bf16_t*)take((size_t)M * E * 2);
    float*  x2  = (float*) take((size_t)M * E * 4);
    bf16_t* h2  = (bf16_t*)take((size_t)M * E * 2);
    bf16_t* ub  = (bf16_t*)take((size_t)M * FF * 2);
    (void)ws_size; (void)in_sizes; (void)n_in; (void)out_size;

    // 1. weights -> bf16, transposed to [out, in]
    convert_transpose<<<4096, 256, 0, stream>>>(wq, wqT, H, E, DH);
    convert_transpose<<<4096, 256, 0, stream>>>(wk, wkT, H, E, DH);
    convert_transpose<<<4096, 256, 0, stream>>>(wv, wvT, H, E, DH);
    convert_transpose<<<4096, 256, 0, stream>>>(wo, woT, 1, E, E);
    convert_transpose<<<4096, 256, 0, stream>>>(w1, w1T, 1, E, FF);
    convert_transpose<<<4096, 256, 0, stream>>>(w2, w2T, 1, FF, E);

    // 2. h1 = rmsnorm(x, g1)
    rmsnorm_kernel<<<M, 256, 0, stream>>>(x, g1, h1, E);

    // 3. QKV projections (bias fused); outputs bf16 [b,t,h*DH+d]
    dim3 gEE(M / 128, E / 64);
    gemm_wmma<<<gEE, 256, 0, stream>>>(h1, wqT, bq, nullptr, nullptr, qb, M, E, E, 0);
    gemm_wmma<<<gEE, 256, 0, stream>>>(h1, wkT, bk, nullptr, nullptr, kb, M, E, E, 0);
    gemm_wmma<<<gEE, 256, 0, stream>>>(h1, wvT, bv, nullptr, nullptr, vb, M, E, E, 0);

    // 4. causal attention
    attn_kernel<<<dim3(Tc / 16, H, Bc), 32, 0, stream>>>(qb, kb, vb, ob, Bc, Tc, H, E, DH);

    // 5. x2 = x + o @ wo + bo
    gemm_wmma<<<gEE, 256, 0, stream>>>(ob, woT, bo, x, x2, nullptr, M, E, E, 0);

    // 6. h2 = rmsnorm(x2, g2)
    rmsnorm_kernel<<<M, 256, 0, stream>>>(x2, g2, h2, E);

    // 7. u = gelu(h2 @ w1 + b1)
    dim3 gEF(M / 128, FF / 64);
    gemm_wmma<<<gEF, 256, 0, stream>>>(h2, w1T, b1, nullptr, nullptr, ub, M, FF, E, 1);

    // 8. out = x2 + u @ w2 + b2
    gemm_wmma<<<gEE, 256, 0, stream>>>(ub, w2T, b2, x2, (float*)d_out, nullptr, M, E, FF, 0);
}
